// VQModule_86517821212613
// MI455X (gfx1250) — matrix operand order repaired and checked by reading, not compiled, not run on hardware
//
#include <hip/hip_runtime.h>
#include <hip/hip_bf16.h>

typedef __attribute__((ext_vector_type(16))) _Float16 v16h;
typedef __attribute__((ext_vector_type(8)))  float    v8f;

#define C_HIDDEN 256
#define KCB      8192
#define NROWS    32768
#define QE_SIZE  (NROWS * C_HIDDEN)   // 8388608
#define SCALE    4096.0f
#define INV2S    (2.0f / SCALE)
#define NS       32                    // codebook columns staged per iteration
#define NSTAGES  (KCB / NS)            // 256
#define BPAD     264                   // padded LDS row stride in f16 (breaks bank conflicts)

struct Frag32 { uint4 a, b; };         // 32 bytes == v16h

static __device__ __forceinline__ v16h frag_cast(Frag32 f) {
  return __builtin_bit_cast(v16h, f);
}

// ---------------------------------------------------------------------------
// Prep: codebook f32 -> f16 (scaled by SCALE) + per-row sum of squares (f32)
// ---------------------------------------------------------------------------
__global__ __launch_bounds__(256) void vq_prep(const float* __restrict__ cb,
                                               _Float16* __restrict__ cbh,
                                               float* __restrict__ csqr) {
  int row = blockIdx.x;
  int t   = threadIdx.x;
  float v = cb[(size_t)row * C_HIDDEN + t];
  cbh[(size_t)row * C_HIDDEN + t] = (_Float16)(v * SCALE);
  float p = v * v;
  #pragma unroll
  for (int m = 16; m >= 1; m >>= 1) p += __shfl_xor(p, m, 32);
  __shared__ float ps[8];
  if ((t & 31) == 0) ps[t >> 5] = p;
  __syncthreads();
  if (t == 0) {
    float s = 0.f;
    #pragma unroll
    for (int i = 0; i < 8; ++i) s += ps[i];
    csqr[row] = s;
  }
}

// Zero the commit-loss accumulator slot every launch (graph-replay safe).
__global__ void vq_zero(float* __restrict__ out) { out[QE_SIZE] = 0.0f; }

// ---------------------------------------------------------------------------
// Main fused kernel: GEMM (f16 WMMA, f32 acc) + argmin + gather + loss.
//   grid = 256 blocks, 256 threads (8 waves); wave w owns rows [blk*128+w*16).
//   A operand lives in registers; B tiles staged in double-buffered LDS and
//   shared by all 8 waves; global prefetch pipelined across the stage loop.
// ---------------------------------------------------------------------------
__global__ __launch_bounds__(256) void vq_main(const float* __restrict__ x,
                                               const float* __restrict__ cb,
                                               const _Float16* __restrict__ cbh,
                                               const float* __restrict__ csqr,
                                               float* __restrict__ out) {
  __shared__ _Float16 bt[2][NS][BPAD];   // 2 x 32 x 264 f16 = 33792 B
  __shared__ float zsq[128];
  __shared__ float dminL[128];
  __shared__ int   idxL[128];
  __shared__ float lossAcc;

  const int t    = threadIdx.x;
  const int lane = t & 31;
  const int wave = t >> 5;
  const int row0 = blockIdx.x * 128;

  if (t == 0) lossAcc = 0.0f;

  // ----- Load A (this wave's 16 z-rows) into registers with ISA A-layout ---
  // lane holds row m = lane&15; khalf = 8*(lane>>4):
  //   frag j covers K = 32j + {khalf..khalf+7} U {16+khalf..16+khalf+7}
  const int mrow   = wave * 16 + (lane & 15);
  const int khalfA = (lane >> 4) * 8;
  const int kbaseB = (lane >> 4) * 8;
  const int ncol   = lane & 15;

  const float4* xp = (const float4*)(x + (size_t)(row0 + mrow) * C_HIDDEN);

  v16h afr[8];
  float ss = 0.f;
  #pragma unroll
  for (int j = 0; j < 8; ++j) {
    const int k0 = j * 32;
    float4 p0 = xp[(k0 + khalfA) >> 2];
    float4 p1 = xp[((k0 + khalfA) >> 2) + 1];
    float4 p2 = xp[(k0 + 16 + khalfA) >> 2];
    float4 p3 = xp[((k0 + 16 + khalfA) >> 2) + 1];
    ss += p0.x*p0.x + p0.y*p0.y + p0.z*p0.z + p0.w*p0.w;
    ss += p1.x*p1.x + p1.y*p1.y + p1.z*p1.z + p1.w*p1.w;
    ss += p2.x*p2.x + p2.y*p2.y + p2.z*p2.z + p2.w*p2.w;
    ss += p3.x*p3.x + p3.y*p3.y + p3.z*p3.z + p3.w*p3.w;
    v16h h;
    h[0]=(_Float16)p0.x; h[1]=(_Float16)p0.y; h[2]=(_Float16)p0.z; h[3]=(_Float16)p0.w;
    h[4]=(_Float16)p1.x; h[5]=(_Float16)p1.y; h[6]=(_Float16)p1.z; h[7]=(_Float16)p1.w;
    h[8]=(_Float16)p2.x;  h[9]=(_Float16)p2.y;  h[10]=(_Float16)p2.z; h[11]=(_Float16)p2.w;
    h[12]=(_Float16)p3.x; h[13]=(_Float16)p3.y; h[14]=(_Float16)p3.z; h[15]=(_Float16)p3.w;
    afr[j] = h;
  }
  // lane and lane^16 together cover the full row -> exact f32 ||z||^2
  float ss2 = ss + __shfl_xor(ss, 16, 32);
  if (lane < 16) zsq[wave * 16 + (lane & 15)] = ss2;

  // ----- Cooperative B staging helpers ------------------------------------
  // Stage tile = NS rows (codebook cols) x 256 f16. 8 threads per row,
  // 64 B (4 x b128) per thread, contiguous in both global and LDS.
  const int srow = t >> 3;           // 0..31
  const int soff = (t & 7) * 32;     // f16 offset within row (64 B chunks)

  uint4 pf0, pf1, pf2, pf3;
  {
    const uint4* p = (const uint4*)(cbh + (size_t)srow * C_HIDDEN + soff);
    pf0 = p[0]; pf1 = p[1]; pf2 = p[2]; pf3 = p[3];
    uint4* dst = (uint4*)(&bt[0][srow][soff]);
    dst[0] = pf0; dst[1] = pf1; dst[2] = pf2; dst[3] = pf3;
  }
  __syncthreads();

  // ----- Stage loop over codebook -----------------------------------------
  float minv[8]; int mini[8];
  #pragma unroll
  for (int r = 0; r < 8; ++r) { minv[r] = 3.4e38f; mini[r] = 0; }

  for (int s = 0; s < NSTAGES; ++s) {
    // Issue next stage's global loads early (latency hidden by compute).
    if (s + 1 < NSTAGES) {
      const uint4* p = (const uint4*)(cbh +
          ((size_t)(s + 1) * NS + srow) * C_HIDDEN + soff);
      pf0 = p[0]; pf1 = p[1]; pf2 = p[2]; pf3 = p[3];
    }

    // Hoist csqr loads for both 16-column chunks of this stage.
    float cs0 = csqr[s * NS + ncol];
    float cs1 = csqr[s * NS + 16 + ncol];

    #pragma unroll
    for (int c = 0; c < 2; ++c) {
      const int n0 = s * NS + c * 16;
      const _Float16* bb = &bt[s & 1][c * 16 + ncol][kbaseB];
      v8f acc0 = {}, acc1 = {};
      #pragma unroll
      for (int j = 0; j < 8; j += 2) {
        const uint4* bp0 = (const uint4*)(bb + j * 32);
        const uint4* bp1 = (const uint4*)(bb + (j + 1) * 32);
        Frag32 f0; f0.a = bp0[0]; f0.b = bp0[2];
        Frag32 f1; f1.a = bp1[0]; f1.b = bp1[2];
        acc0 = __builtin_amdgcn_wmma_f32_16x16x32_f16(false, afr[j], false,
                 frag_cast(f0), (short)0, acc0, false, false);
        acc1 = __builtin_amdgcn_wmma_f32_16x16x32_f16(false, afr[j + 1], false,
                 frag_cast(f1), (short)0, acc1, false, false);
      }
      const float cs = c ? cs1 : cs0;
      const int   n  = n0 + ncol;
      // C layout: lane holds col n, VGPR r holds row m = r + 8*(lane>>4)
      #pragma unroll
      for (int r = 0; r < 8; ++r) {
        float d = cs - INV2S * (acc0[r] + acc1[r]);
        if (d < minv[r]) { minv[r] = d; mini[r] = n; }
      }
    }

    // Commit the prefetched tile to the other buffer, then stage barrier.
    if (s + 1 < NSTAGES) {
      uint4* dst = (uint4*)(&bt[(s + 1) & 1][srow][soff]);
      dst[0] = pf0; dst[1] = pf1; dst[2] = pf2; dst[3] = pf3;
    }
    __syncthreads();
  }

  // ----- Cross-lane (16-wide) min+argmin reduce, first-index tie-break -----
  #pragma unroll
  for (int m = 1; m < 16; m <<= 1) {
    #pragma unroll
    for (int r = 0; r < 8; ++r) {
      float ov = __shfl_xor(minv[r], m, 32);
      int   oi = __shfl_xor(mini[r], m, 32);
      if (ov < minv[r] || (ov == minv[r] && oi < mini[r])) {
        minv[r] = ov; mini[r] = oi;
      }
    }
  }
  if ((lane & 15) == 0) {
    int g = lane >> 4;
    #pragma unroll
    for (int r = 0; r < 8; ++r) {
      int rl = wave * 16 + g * 8 + r;
      dminL[rl] = minv[r];
      idxL[rl]  = mini[r];
    }
  }
  __syncthreads();

  // ----- Output: qe gather (f32 codebook), indices, loss -------------------
  {
    int rl   = t >> 1;
    int half = t & 1;
    int rg   = row0 + rl;
    int idx  = idxL[rl];
    const float4* cp = (const float4*)(cb + (size_t)idx * C_HIDDEN + half * 128);
    float4*       qp = (float4*)(out + (size_t)rg * C_HIDDEN + half * 128);
    #pragma unroll
    for (int i = 0; i < 32; ++i) qp[i] = cp[i];
    if (half == 0) {
      out[QE_SIZE + 1 + rg] = (float)idx;           // indices (as f32)
      atomicAdd(&lossAcc, zsq[rl] + dminL[rl]);     // ||z - zq||^2 per row
    }
  }
  __syncthreads();
  if (t == 0) atomicAdd(out + QE_SIZE, lossAcc * (1.0f / (float)QE_SIZE));
}

// ---------------------------------------------------------------------------
extern "C" void kernel_launch(void* const* d_in, const int* in_sizes, int n_in,
                              void* d_out, int out_size, void* d_ws, size_t ws_size,
                              hipStream_t stream) {
  const float* x  = (const float*)d_in[0];   // [8,4096,256] f32
  const float* cb = (const float*)d_in[1];   // [8192,256] f32

  _Float16* cbh  = (_Float16*)d_ws;                                    // 4 MiB
  float*    csqr = (float*)((char*)d_ws + (size_t)KCB * C_HIDDEN * 2); // 32 KiB
  float*    out  = (float*)d_out;

  vq_prep<<<KCB, 256, 0, stream>>>(cb, cbh, csqr);
  vq_zero<<<1, 1, 0, stream>>>(out);
  vq_main<<<NROWS / 128, 256, 0, stream>>>(x, cb, cbh, csqr, out);
}
